// TransformerBlock_39883066310713
// MI455X (gfx1250) — compile-verified, matrix operand order checked by reading, not run on hardware
//
#include <hip/hip_runtime.h>
#include <hip/hip_bf16.h>
#include <stdint.h>

// ---------------------------------------------------------------------------
// TransformerBlock for MI455X (gfx1250): bf16 WMMA GEMMs, async-to-LDS B
// staging (ASYNCcnt), double-buffered LDS with one barrier per K-step.
// ---------------------------------------------------------------------------

typedef __attribute__((ext_vector_type(16))) __bf16 v16bf;
typedef __attribute__((ext_vector_type(8)))  float  v8f;

#define D_MODEL 1024
#define D_HID   4096
#define SEQ     2048
#define BATCHN  4
#define MROWS   (BATCHN * SEQ)   // 8192

#define BM 128
#define BN 128
#define BK 32
#define LDT 40   // LDS row pitch in bf16 elems (80B: 16B aligned, bank-skewed)

__device__ __forceinline__ unsigned int pack2_bf16(float a, float b) {
  unsigned int ua = __float_as_uint(a); ua += 0x7FFFu + ((ua >> 16) & 1u);
  unsigned int ub = __float_as_uint(b); ub += 0x7FFFu + ((ub >> 16) & 1u);
  return (ua >> 16) | (ub & 0xFFFF0000u);
}
__device__ __forceinline__ unsigned short f2bf(float a) {
  unsigned int u = __float_as_uint(a); u += 0x7FFFu + ((u >> 16) & 1u);
  return (unsigned short)(u >> 16);
}

// 16B global -> LDS async copy (GLOBAL_LOAD_ASYNC_TO_LDS_B128, ASYNCcnt).
// vdst = 32-bit LDS byte address, vaddr = 64-bit global address, GV mode.
__device__ __forceinline__ void async_copy16(const void* g, void* l) {
  asm volatile("global_load_async_to_lds_b128 %0, %1, off"
               :: "v"((unsigned)(uintptr_t)l),
                  "v"((unsigned long long)(uintptr_t)g)
               : "memory");
}
__device__ __forceinline__ void wait_async0() {
  asm volatile("s_wait_asynccnt 0x0" ::: "memory");
}

union FragU { uint4 q[2]; v16bf v; };

// C[M,N] = A[M,K](f32, row-major) x Bt[N,K](bf16, row-major, i.e. B^T)
// Optional epilogue: +bias[N], relu, +resid[M,N]. Dims are tile multiples.
__global__ __launch_bounds__(256) void tf_gemm_bf16wmma(
    const float* __restrict__ A, const unsigned short* __restrict__ Bt,
    float* __restrict__ C, int M, int N, int K,
    long long sA_, long long sB_, long long sC_,
    const float* __restrict__ bias, const float* __restrict__ resid,
    long long sR_, int relu)
{
  __shared__ __align__(16) unsigned short shA[2][BM * LDT];
  __shared__ __align__(16) unsigned short shB[2][BN * LDT];

  A  += (size_t)blockIdx.z * sA_;
  Bt += (size_t)blockIdx.z * sB_;
  C  += (size_t)blockIdx.z * sC_;
  if (resid) resid += (size_t)blockIdx.z * sR_;

  const int tid  = threadIdx.x;
  const int m0   = blockIdx.y * BM;
  const int n0   = blockIdx.x * BN;
  const int lane = tid & 31;
  const int wave = tid >> 5;
  const int wm   = (wave >> 1) * 32;   // 4 waves along M
  const int wn   = (wave & 1) * 64;    // 2 waves along N
  const int l    = lane & 15;
  const int hi   = lane >> 4;

  const int sr   = tid >> 1;           // staging row 0..127
  const int sc   = (tid & 1) * 16;     // staging col (elements)
  const int sidx = sr * LDT + sc;      // this thread's LDS staging slot

  const v8f vzero = {0.f,0.f,0.f,0.f,0.f,0.f,0.f,0.f};
  v8f acc[2][4];
  #pragma unroll
  for (int i = 0; i < 2; ++i)
    #pragma unroll
    for (int j = 0; j < 4; ++j) acc[i][j] = vzero;

  const float*          agp = A  + (size_t)(m0 + sr) * K + sc;
  const unsigned short* bgp = Bt + (size_t)(n0 + sr) * K + sc;

  // prologue: async B tile 0 -> shB[0]; A tile 0 -> registers
  async_copy16(bgp,     &shB[0][sidx]);
  async_copy16(bgp + 8, &shB[0][sidx + 8]);
  float4 ra0, ra1, ra2, ra3;
  {
    const float4* ap = (const float4*)agp;
    ra0 = ap[0]; ra1 = ap[1]; ra2 = ap[2]; ra3 = ap[3];
  }

  int buf = 0;
  for (int k0 = 0;;) {
    // A registers -> bf16 -> shA[buf]
    uint4 pa0, pa1;
    pa0.x = pack2_bf16(ra0.x, ra0.y); pa0.y = pack2_bf16(ra0.z, ra0.w);
    pa0.z = pack2_bf16(ra1.x, ra1.y); pa0.w = pack2_bf16(ra1.z, ra1.w);
    pa1.x = pack2_bf16(ra2.x, ra2.y); pa1.y = pack2_bf16(ra2.z, ra2.w);
    pa1.z = pack2_bf16(ra3.x, ra3.y); pa1.w = pack2_bf16(ra3.z, ra3.w);
    *(uint4*)&shA[buf][sidx]     = pa0;
    *(uint4*)&shA[buf][sidx + 8] = pa1;

    wait_async0();        // this wave's only outstanding asyncs = current tile
    __syncthreads();      // all waves: shA stored, shB async landed

    k0 += BK;
    const bool more = k0 < K;
    if (more) {
      // issued after the barrier -> cannot race last iteration's readers
      async_copy16(bgp + k0,     &shB[buf ^ 1][sidx]);
      async_copy16(bgp + k0 + 8, &shB[buf ^ 1][sidx + 8]);
      const float4* ap = (const float4*)(agp + k0);
      ra0 = ap[0]; ra1 = ap[1]; ra2 = ap[2]; ra3 = ap[3];
    }

    // fragment loads per documented 16-bit A/B VGPR layout:
    // lanes 0-15: K chunks [0..7],[16..23]; lanes 16-31: [8..15],[24..31]
    v16bf af[2], bfr[4];
    #pragma unroll
    for (int mi = 0; mi < 2; ++mi) {
      FragU u;
      const int base = (wm + mi * 16 + l) * LDT + hi * 8;
      u.q[0] = *(const uint4*)&shA[buf][base];
      u.q[1] = *(const uint4*)&shA[buf][base + 16];
      af[mi] = u.v;
    }
    #pragma unroll
    for (int ni = 0; ni < 4; ++ni) {
      FragU u;
      const int base = (wn + ni * 16 + l) * LDT + hi * 8;
      u.q[0] = *(const uint4*)&shB[buf][base];
      u.q[1] = *(const uint4*)&shB[buf][base + 16];
      bfr[ni] = u.v;
    }

    #pragma unroll
    for (int mi = 0; mi < 2; ++mi)
      #pragma unroll
      for (int ni = 0; ni < 4; ++ni)
        acc[mi][ni] = __builtin_amdgcn_wmma_f32_16x16x32_bf16(
            false, af[mi], false, bfr[ni], (short)0, acc[mi][ni], false, false);

    if (!more) break;
    buf ^= 1;
  }

  // epilogue: C/D layout — lane<16: rows M=vgpr(+0..7), lanes 16-31: M+8
  #pragma unroll
  for (int mi = 0; mi < 2; ++mi) {
    const int rb = m0 + wm + mi * 16 + hi * 8;
    #pragma unroll
    for (int ni = 0; ni < 4; ++ni) {
      const int cc = n0 + wn + ni * 16 + l;
      const float bv = bias ? bias[cc] : 0.0f;
      #pragma unroll
      for (int e = 0; e < 8; ++e) {
        float v = acc[mi][ni][e] + bv;
        if (relu) v = fmaxf(v, 0.0f);
        const size_t idx = (size_t)(rb + e) * N + cc;
        if (resid) v += resid[idx];
        C[idx] = v;
      }
    }
  }
}

// row LayerNorm over D_MODEL=1024, one block per row (256 thr x 4 elems)
__global__ __launch_bounds__(256) void tf_layernorm(
    const float* __restrict__ x, const float* __restrict__ g,
    const float* __restrict__ b, float* __restrict__ y)
{
  __shared__ float r1[256], r2[256];
  const int tid = threadIdx.x;
  const float4 v = ((const float4*)(x + (size_t)blockIdx.x * D_MODEL))[tid];
  r1[tid] = v.x + v.y + v.z + v.w;
  r2[tid] = v.x * v.x + v.y * v.y + v.z * v.z + v.w * v.w;
  __syncthreads();
  for (int off = 128; off > 0; off >>= 1) {
    if (tid < off) { r1[tid] += r1[tid + off]; r2[tid] += r2[tid + off]; }
    __syncthreads();
  }
  const float mu  = r1[0] * (1.0f / D_MODEL);
  const float var = r2[0] * (1.0f / D_MODEL) - mu * mu;
  const float inv = rsqrtf(var + 1e-5f);
  const float4 gv = ((const float4*)g)[tid];
  const float4 bv = ((const float4*)b)[tid];
  float4 o;
  o.x = (v.x - mu) * inv * gv.x + bv.x;
  o.y = (v.y - mu) * inv * gv.y + bv.y;
  o.z = (v.z - mu) * inv * gv.z + bv.z;
  o.w = (v.w - mu) * inv * gv.w + bv.w;
  ((float4*)(y + (size_t)blockIdx.x * D_MODEL))[tid] = o;
}

// in-place scale + key-mask + softmax over rows of length SEQ=2048
__global__ __launch_bounds__(256) void tf_softmax_mask(
    float* __restrict__ sc, const int* __restrict__ mask, float scale)
{
  __shared__ float red[256];
  const int tid = threadIdx.x;
  const int row = blockIdx.x;          // 0..B*S-1
  const int b   = row >> 11;           // row / SEQ
  float* p = sc + (size_t)row * SEQ;
  const int* mr = mask + (size_t)b * SEQ;
  const int base = tid * 8;

  float v[8];
  {
    const float4 u0 = ((const float4*)(p + base))[0];
    const float4 u1 = ((const float4*)(p + base))[1];
    v[0]=u0.x; v[1]=u0.y; v[2]=u0.z; v[3]=u0.w;
    v[4]=u1.x; v[5]=u1.y; v[6]=u1.z; v[7]=u1.w;
  }
  int mm[8];
  {
    const int4 m0 = ((const int4*)(mr + base))[0];
    const int4 m1 = ((const int4*)(mr + base))[1];
    mm[0]=m0.x; mm[1]=m0.y; mm[2]=m0.z; mm[3]=m0.w;
    mm[4]=m1.x; mm[5]=m1.y; mm[6]=m1.z; mm[7]=m1.w;
  }
  float mx = -1e30f;
  #pragma unroll
  for (int i = 0; i < 8; ++i) {
    float t = v[i] * scale;
    if (mm[i] == 0) t = -1e9f;
    v[i] = t;
    mx = fmaxf(mx, t);
  }
  red[tid] = mx; __syncthreads();
  for (int off = 128; off > 0; off >>= 1) {
    if (tid < off) red[tid] = fmaxf(red[tid], red[tid + off]);
    __syncthreads();
  }
  mx = red[0];
  __syncthreads();
  float s = 0.f;
  #pragma unroll
  for (int i = 0; i < 8; ++i) { v[i] = __expf(v[i] - mx); s += v[i]; }
  red[tid] = s; __syncthreads();
  for (int off = 128; off > 0; off >>= 1) {
    if (tid < off) red[tid] += red[tid + off];
    __syncthreads();
  }
  const float inv = 1.0f / red[0];
  float4 o0, o1;
  o0.x=v[0]*inv; o0.y=v[1]*inv; o0.z=v[2]*inv; o0.w=v[3]*inv;
  o1.x=v[4]*inv; o1.y=v[5]*inv; o1.z=v[6]*inv; o1.w=v[7]*inv;
  ((float4*)(p + base))[0] = o0;
  ((float4*)(p + base))[1] = o1;
}

// flat f32 -> bf16 (4 elems/thread)
__global__ __launch_bounds__(256) void tf_f32_to_bf16(
    const float* __restrict__ in, unsigned short* __restrict__ out)
{
  const size_t i = (size_t)blockIdx.x * 256 + threadIdx.x;
  const float4 v = ((const float4*)in)[i];
  uint2 o; o.x = pack2_bf16(v.x, v.y); o.y = pack2_bf16(v.z, v.w);
  ((uint2*)out)[i] = o;
}

// f32 [R][C] -> bf16 [C][R], batched via grid.z
__global__ __launch_bounds__(256) void tf_transpose_bf16(
    const float* __restrict__ in, unsigned short* __restrict__ out,
    int R, int C, long long sIn, long long sOut)
{
  __shared__ float t[32][33];
  in  += (size_t)blockIdx.z * sIn;
  out += (size_t)blockIdx.z * sOut;
  const int c0 = blockIdx.x * 32, r0 = blockIdx.y * 32;
  const int tx = threadIdx.x, ty = threadIdx.y;   // 32 x 8
  for (int i = ty; i < 32; i += 8)
    t[i][tx] = in[(size_t)(r0 + i) * C + c0 + tx];
  __syncthreads();
  for (int i = ty; i < 32; i += 8)
    out[(size_t)(c0 + i) * R + r0 + tx] = f2bf(t[tx][i]);
}

extern "C" void kernel_launch(void* const* d_in, const int* in_sizes, int n_in,
                              void* d_out, int out_size, void* d_ws, size_t ws_size,
                              hipStream_t stream) {
  (void)in_sizes; (void)n_in; (void)out_size; (void)ws_size;
  const float* x   = (const float*)d_in[0];
  const int*   msk = (const int*)  d_in[1];
  const float* Wq  = (const float*)d_in[2];
  const float* Wk  = (const float*)d_in[3];
  const float* Wv  = (const float*)d_in[4];
  const float* Wo  = (const float*)d_in[5];
  const float* g1  = (const float*)d_in[6];
  const float* b1  = (const float*)d_in[7];
  const float* g2  = (const float*)d_in[8];
  const float* b2  = (const float*)d_in[9];
  const float* W1  = (const float*)d_in[10];
  const float* bf1 = (const float*)d_in[11];
  const float* W2  = (const float*)d_in[12];
  const float* bf2 = (const float*)d_in[13];

  float* outp = (float*)d_out;
  float* nwp  = outp + (size_t)MROWS * D_MODEL;   // nw lives in d_out

  char* ws = (char*)d_ws;
  const size_t SZ_MD = (size_t)MROWS * D_MODEL * 4;   // 32MB activation slab
  float* xn = (float*)ws;
  char*  rgn = ws + SZ_MD;                            // 128MB reusable region
  float* q  = (float*)rgn;
  float* km = (float*)(rgn + SZ_MD);
  float* vm = (float*)(rgn + 2 * SZ_MD);
  unsigned short* kb = (unsigned short*)(rgn + 3 * SZ_MD);
  unsigned short* vt = (unsigned short*)(rgn + 3 * SZ_MD + SZ_MD / 2);
  float* ff1  = (float*)rgn;                          // reuse after attention
  float* h    = (float*)(ws + 5 * SZ_MD);
  float* hres = (float*)(ws + 6 * SZ_MD);
  float* hn   = xn;                                   // reuse (xn dead by then)
  char* wsw = ws + 7 * SZ_MD;
  const size_t WSZ = (size_t)D_MODEL * D_MODEL * 2;   // 2MB per square weight
  unsigned short* wq_t = (unsigned short*)(wsw);
  unsigned short* wk_t = (unsigned short*)(wsw + WSZ);
  unsigned short* wv_t = (unsigned short*)(wsw + 2 * WSZ);
  unsigned short* wo_t = (unsigned short*)(wsw + 3 * WSZ);
  unsigned short* w1_t = (unsigned short*)(wsw + 4 * WSZ);
  unsigned short* w2_t = (unsigned short*)(wsw + 4 * WSZ + (size_t)D_MODEL * D_HID * 2);

  const dim3 tb(256), t32(32, 8);

  // weight transposes to bf16 [N][K]
  tf_transpose_bf16<<<dim3(32, 32, 1), t32, 0, stream>>>(Wq, wq_t, D_MODEL, D_MODEL, 0, 0);
  tf_transpose_bf16<<<dim3(32, 32, 1), t32, 0, stream>>>(Wk, wk_t, D_MODEL, D_MODEL, 0, 0);
  tf_transpose_bf16<<<dim3(32, 32, 1), t32, 0, stream>>>(Wv, wv_t, D_MODEL, D_MODEL, 0, 0);
  tf_transpose_bf16<<<dim3(32, 32, 1), t32, 0, stream>>>(Wo, wo_t, D_MODEL, D_MODEL, 0, 0);
  tf_transpose_bf16<<<dim3(128, 32, 1), t32, 0, stream>>>(W1, w1_t, D_MODEL, D_HID, 0, 0);
  tf_transpose_bf16<<<dim3(32, 128, 1), t32, 0, stream>>>(W2, w2_t, D_HID, D_MODEL, 0, 0);

  // LN1
  tf_layernorm<<<MROWS, tb, 0, stream>>>(x, g1, b1, xn);

  // Q, K, V projections
  tf_gemm_bf16wmma<<<dim3(D_MODEL / BN, MROWS / BM, 1), tb, 0, stream>>>(
      xn, wq_t, q, MROWS, D_MODEL, D_MODEL, 0, 0, 0, nullptr, nullptr, 0, 0);
  tf_gemm_bf16wmma<<<dim3(D_MODEL / BN, MROWS / BM, 1), tb, 0, stream>>>(
      xn, wk_t, km, MROWS, D_MODEL, D_MODEL, 0, 0, 0, nullptr, nullptr, 0, 0);
  tf_gemm_bf16wmma<<<dim3(D_MODEL / BN, MROWS / BM, 1), tb, 0, stream>>>(
      xn, wv_t, vm, MROWS, D_MODEL, D_MODEL, 0, 0, 0, nullptr, nullptr, 0, 0);

  // K -> bf16 (scores B operand is already [N=S][K=D] row-major)
  tf_f32_to_bf16<<<(MROWS * D_MODEL / 4) / 256, tb, 0, stream>>>(km, kb);
  // V -> bf16 transposed per batch: [D][S]
  tf_transpose_bf16<<<dim3(D_MODEL / 32, SEQ / 32, BATCHN), t32, 0, stream>>>(
      vm, vt, SEQ, D_MODEL, (long long)SEQ * D_MODEL, (long long)D_MODEL * SEQ);

  // scores = Q K^T, straight into the nw region of d_out
  tf_gemm_bf16wmma<<<dim3(SEQ / BN, SEQ / BM, BATCHN), tb, 0, stream>>>(
      q, kb, nwp, SEQ, SEQ, D_MODEL,
      (long long)SEQ * D_MODEL, (long long)SEQ * D_MODEL, (long long)SEQ * SEQ,
      nullptr, nullptr, 0, 0);

  // scale + mask + softmax in place -> nw
  tf_softmax_mask<<<MROWS, tb, 0, stream>>>(nwp, msk, 0.03125f);

  // h = nw @ V
  tf_gemm_bf16wmma<<<dim3(D_MODEL / BN, SEQ / BM, BATCHN), tb, 0, stream>>>(
      nwp, vt, h, SEQ, D_MODEL, SEQ,
      (long long)SEQ * SEQ, (long long)D_MODEL * SEQ, (long long)SEQ * D_MODEL,
      nullptr, nullptr, 0, 0);

  // hres = h @ Wo + x
  tf_gemm_bf16wmma<<<dim3(D_MODEL / BN, MROWS / BM, 1), tb, 0, stream>>>(
      h, wo_t, hres, MROWS, D_MODEL, D_MODEL, 0, 0, 0, nullptr, x, 0, 0);

  // LN2
  tf_layernorm<<<MROWS, tb, 0, stream>>>(hres, g2, b2, hn);

  // ff1 = relu(hn @ W1 + b1)
  tf_gemm_bf16wmma<<<dim3(D_HID / BN, MROWS / BM, 1), tb, 0, stream>>>(
      hn, w1_t, ff1, MROWS, D_HID, D_MODEL, 0, 0, 0, bf1, nullptr, 0, 1);

  // out = ff1 @ W2 + b2 + hres
  tf_gemm_bf16wmma<<<dim3(D_MODEL / BN, MROWS / BM, 1), tb, 0, stream>>>(
      ff1, w2_t, outp, MROWS, D_MODEL, D_HID, 0, 0, 0, bf2, hres, 0, 0);
}
